// Attention_21887153340684
// MI455X (gfx1250) — matrix operand order repaired and checked by reading, not compiled, not run on hardware
//
#include <hip/hip_runtime.h>
#include <hip/hip_bf16.h>

// Problem constants (from reference): N=256, L=2048, QK=512, H=128
#define N_ 256
#define L_ 2048
#define QK_ 512
#define H_ 128
#define CHUNK 128                 // L rows per workgroup
#define KT 32                     // K-tile per WMMA step (f16 16x16x32)
#define NCHUNK (L_ / CHUNK)       // 16
#define KSTEPS (QK_ / KT)         // 16

typedef _Float16 half8  __attribute__((ext_vector_type(8)));
typedef _Float16 half16 __attribute__((ext_vector_type(16)));
typedef float    float8 __attribute__((ext_vector_type(8)));
typedef int      v4i_   __attribute__((ext_vector_type(4)));

typedef __attribute__((address_space(1))) v4i_* gptr_v4i;     // global int4*
typedef __attribute__((address_space(3))) v4i_* lptr_v4i;     // LDS int4*

// Async copy global->LDS (gfx1250 GLOBAL_LOAD_ASYNC_TO_LDS_B128, ASYNCcnt),
// guarded so the file compiles on toolchains without the builtins.
#if defined(__HIP_DEVICE_COMPILE__) &&                                        \
    __has_builtin(__builtin_amdgcn_global_load_async_to_lds_b128) &&          \
    __has_builtin(__builtin_amdgcn_s_wait_asynccnt)
#define USE_ASYNC_LDS 1
#else
#define USE_ASYNC_LDS 0
#endif

// ---------------------------------------------------------------------------
// Kernel 0: convert Ua (H x QK fp32, row-major) to f16 once; reused by all WGs
// ---------------------------------------------------------------------------
__global__ __launch_bounds__(256) void cvt_ua_kernel(const float* __restrict__ Ua,
                                                     _Float16* __restrict__ Uah) {
    int i = blockIdx.x * 256 + threadIdx.x;   // H_*QK_ = 65536 elements
    if (i < H_ * QK_) Uah[i] = (_Float16)Ua[i];
}

// ---------------------------------------------------------------------------
// Kernel 1: q[n][h] = dot(last_hidden[n,:], Wa[h,:])  (tiny: 33 MFLOP, fp32)
// ---------------------------------------------------------------------------
__global__ __launch_bounds__(H_) void qproj_kernel(const float* __restrict__ lh,
                                                   const float* __restrict__ Wa,
                                                   float* __restrict__ q) {
    int n = blockIdx.x;
    int h = threadIdx.x;   // 0..127
    const float4* x4 = (const float4*)(lh + (size_t)n * QK_);
    const float4* w4 = (const float4*)(Wa + (size_t)h * QK_);
    float acc = 0.f;
    #pragma unroll 4
    for (int i = 0; i < QK_ / 4; ++i) {
        float4 a = x4[i], b = w4[i];
        acc += a.x * b.x + a.y * b.y + a.z * b.z + a.w * b.w;
    }
    q[(size_t)n * H_ + h] = acc;
}

// ---------------------------------------------------------------------------
// Kernel 2 (main, fused): per (n, L-chunk of 128):
//   kproj = enc_chunk @ Ua^T via v_wmma_f32_16x16x32_f16 (f16 in, f32 accum)
//   e_l   = sum_h va[h]*tanh(q[n,h] + kproj[l,h])
//   chunk-local softmax stats (max, sum) + weighted context partial
// Grid: (NCHUNK, N), 256 threads = 8 waves (wave32). LDS double-buffered:
// one barrier per k-step; B tile staged via async-to-LDS when available.
// ---------------------------------------------------------------------------
__global__ __launch_bounds__(256) void energies_kernel(
    const float*    __restrict__ enc,     // [N][L][QK] fp32
    const _Float16* __restrict__ Uah,     // [H][QK] f16
    const float*    __restrict__ qn,      // [N][H]
    const float*    __restrict__ va,      // [H]
    float*          __restrict__ eraw,    // [N][L]  raw energies (d_out weights region)
    float*          __restrict__ ctxP,    // [N][NCHUNK][QK] context partials
    float*          __restrict__ cmax,    // [N][NCHUNK]
    float*          __restrict__ csum)    // [N][NCHUNK]
{
    __shared__ __align__(16) _Float16 aLds[2][CHUNK * KT];  // 2 x 8 KB enc tile (f16)
    __shared__ __align__(16) _Float16 bLds[2][H_ * KT];     // 2 x 8 KB Ua tile (f16), [h][k]
    __shared__ float eLds[CHUNK];
    __shared__ float pLds[CHUNK];
    __shared__ float wred[8];

    const int n    = blockIdx.y;
    const int c    = blockIdx.x;
    const int l0   = c * CHUNK;
    const int t    = threadIdx.x;
    const int wave = t >> 5;       // 0..7  (also this wave's M-tile)
    const int lane = t & 31;

    const float* encC = enc + ((size_t)n * L_ + l0) * QK_;

    const float8 fz = {0.f, 0.f, 0.f, 0.f, 0.f, 0.f, 0.f, 0.f};
    float8 acc[8];
    #pragma unroll
    for (int i = 0; i < 8; ++i) acc[i] = fz;

    // Cooperative staging indices (fixed per thread)
    const int sRow = t >> 1;            // 0..127
    const int sCol = (t & 1) * 16;      // 0 or 16

    // in-flight staged A tile (fp32, converted at commit time)
    float4 a0, a1, a2, a3;
#if !USE_ASYNC_LDS
    half8 b0, b1;
#endif

    auto issueLoads = [&](int kk2) {
        const float4* src = (const float4*)(encC + (size_t)sRow * QK_ + kk2 * KT + sCol);
        a0 = src[0]; a1 = src[1]; a2 = src[2]; a3 = src[3];
        const int sb = kk2 & 1;
#if USE_ASYNC_LDS
        _Float16* gsrc = const_cast<_Float16*>(Uah + (size_t)sRow * QK_ + kk2 * KT + sCol);
        _Float16* ldst = &bLds[sb][sRow * KT + sCol];
        __builtin_amdgcn_global_load_async_to_lds_b128(
            (gptr_v4i)gsrc, (lptr_v4i)ldst, 0, 0);
        __builtin_amdgcn_global_load_async_to_lds_b128(
            (gptr_v4i)(gsrc + 8), (lptr_v4i)(ldst + 8), 0, 0);
#else
        const half8* bs = (const half8*)(Uah + (size_t)sRow * QK_ + kk2 * KT + sCol);
        b0 = bs[0]; b1 = bs[1];
        (void)sb;
#endif
    };
    auto commitStage = [&](int kk2) {
        const int sb = kk2 & 1;
        half8 h0 = {(_Float16)a0.x, (_Float16)a0.y, (_Float16)a0.z, (_Float16)a0.w,
                    (_Float16)a1.x, (_Float16)a1.y, (_Float16)a1.z, (_Float16)a1.w};
        half8 h1 = {(_Float16)a2.x, (_Float16)a2.y, (_Float16)a2.z, (_Float16)a2.w,
                    (_Float16)a3.x, (_Float16)a3.y, (_Float16)a3.z, (_Float16)a3.w};
        *(half8*)(&aLds[sb][sRow * KT + sCol])     = h0;
        *(half8*)(&aLds[sb][sRow * KT + sCol + 8]) = h1;
#if !USE_ASYNC_LDS
        *(half8*)(&bLds[sb][sRow * KT + sCol])     = b0;
        *(half8*)(&bLds[sb][sRow * KT + sCol + 8]) = b1;
#endif
    };

    // prologue: stage k-tile 0 into buffer 0
    issueLoads(0);
    commitStage(0);

    for (int kk = 0; kk < KSTEPS; ++kk) {
#if USE_ASYNC_LDS
        __builtin_amdgcn_s_wait_asynccnt(0);   // B tile of this iteration landed in LDS
#endif
        __syncthreads();                       // staging of buf (kk&1) visible to all

        if (kk + 1 < KSTEPS) issueLoads(kk + 1);   // overlap next tile's global loads

        const int cb = kk & 1;
        // --- A fragment (16x32 f16; ISA §7.12.2 layout)
        union { half16 v; half8 h[2]; } afrag;
        {
            const int m     = lane & 15;
            const int row   = wave * 16 + m;
            const int kbase = (lane >> 4) * 8;        // lanes 16-31 hold K+8
            afrag.h[0] = *(const half8*)(&aLds[cb][row * KT + kbase]);        // K kbase..+7
            afrag.h[1] = *(const half8*)(&aLds[cb][row * KT + kbase + 16]);   // K kbase+16..+23
        }
        // --- 8 N-tiles of B (32x16 f16), accumulate
        #pragma unroll
        for (int nt = 0; nt < 8; ++nt) {
            union { half16 v; half8 h[2]; } bfrag;
            const int h  = nt * 16 + (lane & 15);     // column (= hidden idx)
            const int kb = (lane >> 4) * 8;           // lanes 16-31 hold K+8
            bfrag.h[0] = *(const half8*)(&bLds[cb][h * KT + kb]);
            bfrag.h[1] = *(const half8*)(&bLds[cb][h * KT + kb + 16]);
            acc[nt] = __builtin_amdgcn_wmma_f32_16x16x32_f16(
                false, afrag.v, false, bfrag.v, (short)0, acc[nt], false, false);
        }

        if (kk + 1 < KSTEPS) commitStage(kk + 1);  // fill the other buffer
    }

    // --- epilogue: e_l = sum_h va[h]*tanh(q[h] + kproj[l][h])
    // C/D layout: lane holds rows M = r + 8*(lane>=16), column h = nt*16 + lane%16
    float vaL[8], qL[8];
    #pragma unroll
    for (int nt = 0; nt < 8; ++nt) {
        const int h = nt * 16 + (lane & 15);
        vaL[nt] = va[h];
        qL[nt]  = qn[(size_t)n * H_ + h];
    }
    float part[8];
    #pragma unroll
    for (int r = 0; r < 8; ++r) {
        float s = 0.f;
        #pragma unroll
        for (int nt = 0; nt < 8; ++nt)
            s += vaL[nt] * tanhf(qL[nt] + acc[nt][r]);
        part[r] = s;
    }
    // reduce over the 16 lanes of each half-wave (h dimension)
    #pragma unroll
    for (int r = 0; r < 8; ++r) {
        float s = part[r];
        #pragma unroll
        for (int m = 1; m < 16; m <<= 1) s += __shfl_xor(s, m, 16);
        part[r] = s;
    }
    if ((lane & 15) == 0) {
        const int mbase = (lane >> 4) * 8;            // 0 or 8
        #pragma unroll
        for (int r = 0; r < 8; ++r)
            eLds[wave * 16 + mbase + r] = part[r];
    }
    __syncthreads();

    // --- chunk softmax stats over 128 energies
    float v = (t < CHUNK) ? eLds[t] : -3.0e38f;
    #pragma unroll
    for (int m = 16; m >= 1; m >>= 1) v = fmaxf(v, __shfl_xor(v, m, 32));
    if (lane == 0) wred[wave] = v;
    __syncthreads();
    float gm = wred[0];
    #pragma unroll
    for (int i = 1; i < 8; ++i) gm = fmaxf(gm, wred[i]);
    __syncthreads();                                  // before reusing wred

    float p = 0.f;
    if (t < CHUNK) {
        p = __expf(eLds[t] - gm);
        pLds[t] = p;
        eraw[(size_t)n * L_ + l0 + t] = eLds[t];      // raw energies for finalize
    }
    float sv = p;
    #pragma unroll
    for (int m = 16; m >= 1; m >>= 1) sv += __shfl_xor(sv, m, 32);
    if (lane == 0) wred[wave] = sv;
    __syncthreads();
    if (t == 0) {
        float gs = 0.f;
        #pragma unroll
        for (int i = 0; i < 8; ++i) gs += wred[i];
        cmax[(size_t)n * NCHUNK + c] = gm;
        csum[(size_t)n * NCHUNK + c] = gs;
    }

    // --- chunk context partial: ctx[d] = sum_l p_l * enc[l][d]; chunk is cache-hot
    {
        const int d = t * 2;                          // 256 threads x 2 dims = 512
        float c0 = 0.f, c1 = 0.f;
        for (int l = 0; l < CHUNK; ++l) {
            const float pl = pLds[l];
            const float2 ev = *(const float2*)(encC + (size_t)l * QK_ + d);
            c0 += pl * ev.x;
            c1 += pl * ev.y;
        }
        float2* outp = (float2*)(ctxP + ((size_t)n * NCHUNK + c) * QK_ + d);
        *outp = make_float2(c0, c1);
    }
}

// ---------------------------------------------------------------------------
// Kernel 3: merge chunk partials per batch row (log-sum-exp rescale), write
// final attention weights and context vector.
// ---------------------------------------------------------------------------
__global__ __launch_bounds__(256) void finalize_kernel(
    const float* __restrict__ cmax,
    const float* __restrict__ csum,
    const float* __restrict__ ctxP,
    float*       __restrict__ outCtx,    // [N][QK]
    float*       __restrict__ weights)   // [N][L], holds raw energies on entry
{
    __shared__ float sm[NCHUNK], ss[NCHUNK], scale[NCHUNK];
    __shared__ float gstats[2];
    const int n = blockIdx.x;
    const int t = threadIdx.x;

    if (t < NCHUNK) {
        sm[t] = cmax[(size_t)n * NCHUNK + t];
        ss[t] = csum[(size_t)n * NCHUNK + t];
    }
    __syncthreads();
    if (t == 0) {
        float gm = sm[0];
        for (int i = 1; i < NCHUNK; ++i) gm = fmaxf(gm, sm[i]);
        float gs = 0.f;
        for (int i = 0; i < NCHUNK; ++i) {
            scale[i] = __expf(sm[i] - gm);
            gs += ss[i] * scale[i];
        }
        gstats[0] = gm;
        gstats[1] = 1.0f / gs;
    }
    __syncthreads();
    const float gm  = gstats[0];
    const float inv = gstats[1];

    // final softmax weights
    for (int l = t; l < L_; l += 256) {
        const float e = weights[(size_t)n * L_ + l];
        weights[(size_t)n * L_ + l] = __expf(e - gm) * inv;
    }
    // final context
    for (int d = t; d < QK_; d += 256) {
        float a = 0.f;
        #pragma unroll
        for (int c2 = 0; c2 < NCHUNK; ++c2)
            a += ctxP[((size_t)n * NCHUNK + c2) * QK_ + d] * scale[c2];
        outCtx[(size_t)n * QK_ + d] = a * inv;
    }
}

// ---------------------------------------------------------------------------
extern "C" void kernel_launch(void* const* d_in, const int* in_sizes, int n_in,
                              void* d_out, int out_size, void* d_ws, size_t ws_size,
                              hipStream_t stream) {
    const float* last_hidden = (const float*)d_in[0];   // N*1*QK
    const float* enc         = (const float*)d_in[1];   // N*L*QK
    const float* Wa          = (const float*)d_in[2];   // H*QK
    const float* Ua          = (const float*)d_in[3];   // H*QK
    const float* va          = (const float*)d_in[4];   // H

    float* out     = (float*)d_out;
    float* outCtx  = out;                      // [N][QK]
    float* weights = out + (size_t)N_ * QK_;   // [N][L]

    // workspace layout (~8.7 MB total)
    char* ws = (char*)d_ws;
    _Float16* Uah = (_Float16*)ws;                         // 128 KB
    float* q      = (float*)(ws + (size_t)131072);         // 128 KB
    float* cmax   = (float*)(ws + (size_t)262144);         //  16 KB
    float* csum   = (float*)(ws + (size_t)278528);         //  16 KB
    float* ctxP   = (float*)(ws + (size_t)294912);         //   8 MB

    cvt_ua_kernel<<<(H_ * QK_) / 256, 256, 0, stream>>>(Ua, Uah);
    qproj_kernel<<<N_, H_, 0, stream>>>(last_hidden, Wa, q);
    dim3 grid(NCHUNK, N_);
    energies_kernel<<<grid, 256, 0, stream>>>(enc, Uah, q, va, weights, ctxP, cmax, csum);
    finalize_kernel<<<N_, 256, 0, stream>>>(cmax, csum, ctxP, outCtx, weights);
}
